// LSTMBlock_84799834292959
// MI455X (gfx1250) — matrix operand order repaired and checked by reading, not compiled, not run on hardware
//
#include <hip/hip_runtime.h>
#include <hip/hip_bf16.h>
#include <math.h>

#define B_   1024
#define T_   64
#define D_   256
#define H_   512
#define K_   768          // D + H
#define G4_  2048         // 4*H
#define LDA  784          // LDS row stride (ushorts): 768 + 16 pad, keeps 16B align
#define LN_EPS 1e-5f

typedef __attribute__((ext_vector_type(16))) __bf16        v16bf;
typedef __attribute__((ext_vector_type(8)))  float          v8f;
typedef __attribute__((ext_vector_type(8)))  unsigned short v8us;

union ABFrag { v16bf bf; v8us h[2]; unsigned short us[16]; };

__device__ __forceinline__ unsigned short f2bf(float f) {
  unsigned int u = __float_as_uint(f);
  u += 0x7FFFu + ((u >> 16) & 1u);         // round-to-nearest-even
  return (unsigned short)(u >> 16);
}
__device__ __forceinline__ float sigm_(float x) { return 1.0f / (1.0f + __expf(-x)); }

// ---------------- prep kernels ----------------
__global__ void k_prep_w(const float* __restrict__ wih,
                         const float* __restrict__ whh,
                         unsigned short* __restrict__ wcat) {
  int i = blockIdx.x * 256 + threadIdx.x;            // < 2048*768
  int j = i / K_;
  int k = i - j * K_;
  float v = (k < D_) ? wih[j * D_ + k] : whh[j * H_ + (k - D_)];
  wcat[i] = f2bf(v);
}

__global__ void k_prep_x(const float* __restrict__ x,
                         unsigned short* __restrict__ xbf) {
  size_t i = (size_t)blockIdx.x * 256 + threadIdx.x; // < B*T*D
  xbf[i] = f2bf(x[i]);
}

__global__ void k_init(const float* __restrict__ h0, const float* __restrict__ c0,
                       float* __restrict__ hbuf, float* __restrict__ cbuf) {
  size_t i = (size_t)blockIdx.x * 256 + threadIdx.x; // < B*H
  hbuf[i] = h0[i];
  cbuf[i] = c0[i];
}

// ---------------- fused LSTM step ----------------
// grid 256 = (B/32 row blocks) x (H/64 col blocks); block = 128 = 4 wave32.
// Each wave: TWO 16x16 M-tiles x 16 cols of H for all 4 gates (8 accumulators),
// reusing each B fragment across both M-tiles (32 FLOP per L2 byte).
__global__ __launch_bounds__(128)
void k_step(const unsigned short* __restrict__ xbf,
            const unsigned short* __restrict__ wcat,
            const float* __restrict__ hin,  float* __restrict__ hout,
            const float* __restrict__ cin,  float* __restrict__ cout_,
            const float* __restrict__ b_ih, const float* __restrict__ b_hh,
            const int*   __restrict__ dones,
            float* __restrict__ out, int t) {
  __shared__ unsigned short sA[32 * LDA];            // 32 rows x 768 (+pad) bf16
  const int tid = threadIdx.x;
  const int rb  = blockIdx.x >> 3;
  const int cb  = blockIdx.x & 7;
  const int b0  = rb * 32;

  // stage A cols [0,256): x_t (already bf16), 16B vector copies
  for (int c = tid; c < 32 * 32; c += 128) {
    int row = c >> 5;
    int col = (c & 31) * 8;
    const v8us* src = reinterpret_cast<const v8us*>(
        xbf + ((size_t)(b0 + row) * T_ + t) * D_ + col);
    *reinterpret_cast<v8us*>(&sA[row * LDA + col]) = *src;
  }
  // stage A cols [256,768): mask * h_prev, f32 -> bf16
  for (int c = tid; c < 32 * 64; c += 128) {
    int row = c >> 6;
    int col = (c & 63) * 8;
    int b   = b0 + row;
    float mask = (t == 0) ? 1.0f : (1.0f - (float)dones[b * T_ + (t - 1)]);
    const float4* s = reinterpret_cast<const float4*>(hin + (size_t)b * H_ + col);
    float4 p0 = s[0], p1 = s[1];
    v8us d;
    d[0] = f2bf(p0.x * mask); d[1] = f2bf(p0.y * mask);
    d[2] = f2bf(p0.z * mask); d[3] = f2bf(p0.w * mask);
    d[4] = f2bf(p1.x * mask); d[5] = f2bf(p1.y * mask);
    d[6] = f2bf(p1.z * mask); d[7] = f2bf(p1.w * mask);
    *reinterpret_cast<v8us*>(&sA[row * LDA + D_ + col]) = d;
  }
  __syncthreads();

  const int wave  = tid >> 5;
  const int lane  = tid & 31;
  const int n     = lane & 15;        // N index (and A row M index within tile)
  const int khalf = lane >> 4;        // 0/1: K-half selector
  const int j0    = cb * 64 + wave * 16;
  const int j     = j0 + n;

  float bias[4];
#pragma unroll
  for (int q = 0; q < 4; ++q) bias[q] = b_ih[q * H_ + j] + b_hh[q * H_ + j];

  v8f acc[2][4] = {};                  // [M-tile][gate i,f,g,o] fp32 accumulators

  const int koffA = khalf * 8;
  for (int kk = 0; kk < K_; kk += 32) {
    // Two A fragments (rows n and n+16 of the 32-row block), 16x32 bf16 each
    ABFrag a0, a1;
    const unsigned short* ap0 = &sA[n * LDA + kk + koffA];
    const unsigned short* ap1 = &sA[(n + 16) * LDA + kk + koffA];
    a0.h[0] = *reinterpret_cast<const v8us*>(ap0);
    a0.h[1] = *reinterpret_cast<const v8us*>(ap0 + 16);
    a1.h[0] = *reinterpret_cast<const v8us*>(ap1);
    a1.h[1] = *reinterpret_cast<const v8us*>(ap1 + 16);
    if (kk + 32 < K_)  // pull next K-slice of W row toward the WGP
      __builtin_prefetch(wcat + (size_t)j * K_ + kk + 32, 0, 1);
#pragma unroll
    for (int q = 0; q < 4; ++q) {
      // B fragment 32x16 bf16: row (q*H + j) of Wcat, K-half by lane group.
      // Loaded ONCE, consumed by both M-tiles.
      ABFrag bm;
      const unsigned short* bp =
          wcat + (size_t)(q * H_ + j) * K_ + kk + khalf * 8;
      bm.h[0] = *reinterpret_cast<const v8us*>(bp);
      bm.h[1] = *reinterpret_cast<const v8us*>(bp + 16);
      acc[0][q] = __builtin_amdgcn_wmma_f32_16x16x32_bf16(
          false, a0.bf, false, bm.bf, (short)0, acc[0][q], false, false);
      acc[1][q] = __builtin_amdgcn_wmma_f32_16x16x32_bf16(
          false, a1.bf, false, bm.bf, (short)0, acc[1][q], false, false);
    }
  }

  // epilogue: gates -> cell update -> h, one (b,j) per lane-row per tile
#pragma unroll
  for (int tile = 0; tile < 2; ++tile) {
#pragma unroll
    for (int r = 0; r < 8; ++r) {
      int m = tile * 16 + r + khalf * 8;
      int b = b0 + m;
      float mask  = (t == 0) ? 1.0f : (1.0f - (float)dones[b * T_ + (t - 1)]);
      float cprev = cin[(size_t)b * H_ + j] * mask;
      float gi = acc[tile][0][r] + bias[0];
      float gf = acc[tile][1][r] + bias[1];
      float gg = acc[tile][2][r] + bias[2];
      float go = acc[tile][3][r] + bias[3];
      float cnew = sigm_(gf) * cprev + sigm_(gi) * tanhf(gg);
      float hnew = sigm_(go) * tanhf(cnew);
      cout_[(size_t)b * H_ + j] = cnew;
      hout [(size_t)b * H_ + j] = hnew;
      out[((size_t)b * T_ + t) * H_ + j] = hnew;
    }
  }
}

// ---------------- LayerNorm (in place over d_out rows) ----------------
__global__ __launch_bounds__(128)
void k_ln(float* __restrict__ out, const float* __restrict__ gamma,
          const float* __restrict__ beta) {
  const int lane = threadIdx.x & 31;
  const int row  = blockIdx.x * 4 + (threadIdx.x >> 5);  // < B*T
  float* p = out + (size_t)row * H_;
  float v[16];
  float s = 0.f, sq = 0.f;
#pragma unroll
  for (int i = 0; i < 16; ++i) {
    v[i] = p[lane + i * 32];
    s += v[i];
    sq += v[i] * v[i];
  }
#pragma unroll
  for (int off = 16; off > 0; off >>= 1) {
    s  += __shfl_xor(s,  off, 32);
    sq += __shfl_xor(sq, off, 32);
  }
  float mu  = s * (1.0f / H_);
  float var = sq * (1.0f / H_) - mu * mu;
  float rs  = rsqrtf(var + LN_EPS);
#pragma unroll
  for (int i = 0; i < 16; ++i) {
    int col = lane + i * 32;
    p[col] = gamma[col] * (v[i] - mu) * rs + beta[col];
  }
}

// ---------------- launch ----------------
extern "C" void kernel_launch(void* const* d_in, const int* in_sizes, int n_in,
                              void* d_out, int out_size, void* d_ws, size_t ws_size,
                              hipStream_t stream) {
  const float* x     = (const float*)d_in[0];
  const float* h0    = (const float*)d_in[1];
  const float* c0    = (const float*)d_in[2];
  const float* W_ih  = (const float*)d_in[3];
  const float* W_hh  = (const float*)d_in[4];
  const float* b_ih  = (const float*)d_in[5];
  const float* b_hh  = (const float*)d_in[6];
  const float* gamma = (const float*)d_in[7];
  const float* beta  = (const float*)d_in[8];
  const int*   dones = (const int*)d_in[9];
  float* out = (float*)d_out;

  // workspace layout
  char* ws = (char*)d_ws;
  unsigned short* wcat = (unsigned short*)ws;                      //  3 MB
  unsigned short* xbf  = (unsigned short*)(ws + (3ull << 20));     // 32 MB
  float* hbuf = (float*)(ws + (35ull << 20));                      //  4 MB (x2 bufs)
  float* cbuf = (float*)(ws + (39ull << 20));                      //  4 MB (x2 bufs)
  const size_t BH = (size_t)B_ * H_;

  k_prep_w<<<(G4_ * K_) / 256, 256, 0, stream>>>(W_ih, W_hh, wcat);
  k_prep_x<<<((size_t)B_ * T_ * D_) / 256, 256, 0, stream>>>(x, xbf);
  k_init<<<BH / 256, 256, 0, stream>>>(h0, c0, hbuf, cbuf);

  for (int t = 0; t < T_; ++t) {
    float* hin   = hbuf + (size_t)(t & 1) * BH;
    float* hout  = hbuf + (size_t)((t + 1) & 1) * BH;
    float* cin   = cbuf + (size_t)(t & 1) * BH;
    float* coutp = cbuf + (size_t)((t + 1) & 1) * BH;
    k_step<<<256, 128, 0, stream>>>(xbf, wcat, hin, hout, cin, coutp,
                                    b_ih, b_hh, dones, out, t);
  }

  k_ln<<<(B_ * T_) / 4, 128, 0, stream>>>(out, gamma, beta);

  // tuple tail: h_n then c_n (final state lands in buffer 0 after 64 steps)
  hipMemcpyAsync(out + (size_t)B_ * T_ * H_,       hbuf, BH * sizeof(float),
                 hipMemcpyDeviceToDevice, stream);
  hipMemcpyAsync(out + (size_t)B_ * T_ * H_ + BH,  cbuf, BH * sizeof(float),
                 hipMemcpyDeviceToDevice, stream);
}
